// GINetwork_14173392077525
// MI455X (gfx1250) — matrix operand order repaired and checked by reading, not compiled, not run on hardware
//
#include <hip/hip_runtime.h>
#include <math.h>

// ---------------- problem constants (match reference) ----------------
#define S_    8
#define N_    50000
#define M_    128
#define DIN_  32
#define H_    128
#define DOUT_ 1

#define KP0   64     // padded K for layer-0 weights (din0 = 33)
#define KP1   160    // padded K for layer-1/2 weights (din1 = din2 = 129)
#define PUSTR 160    // phiU row stride (padded), also fused-tile row stride

typedef __attribute__((ext_vector_type(16))) _Float16 v16h;
typedef __attribute__((ext_vector_type(8)))  float    v8f;

// precise ELU (setup path, f32 weights)
__device__ __forceinline__ float eluf(float x) { return x > 0.f ? x : expm1f(x); }

// branchless fast ELU (fused forward path; result is rounded to f16 anyway).
// __expf -> v_exp_f32; select -> v_cndmask. No exec-mask branches.
__device__ __forceinline__ float eluf_fast(float x) {
  float e = __expf(fminf(x, 0.f)) - 1.f;
  return x > 0.f ? x : e;
}

__device__ __forceinline__ float block_reduce(float v, float* red, int tid, int T) {
  red[tid] = v; __syncthreads();
  for (int off = T >> 1; off > 0; off >>= 1) {
    if (tid < off) red[tid] += red[tid + off];
    __syncthreads();
  }
  float r = red[0]; __syncthreads();
  return r;
}

// ---------------- tiny init: zero the KL accumulators ----------------
__global__ void zero_kl_kernel(float* kl) {
  if (threadIdx.x < S_) kl[threadIdx.x] = 0.f;
}

// ---------------- phiU0 = elu(append_ones(inducing_points)), padded ----------------
__global__ void build_phiU0_kernel(const float* __restrict__ ip, float* __restrict__ phiU) {
  int idx = blockIdx.x * blockDim.x + threadIdx.x;
  if (idx >= S_ * M_ * PUSTR) return;
  int j = idx % PUSTR;
  int m = (idx / PUSTR) % M_;
  float v = 0.f;
  if (j < DIN_)       v = eluf(ip[m * DIN_ + j]);
  else if (j == DIN_) v = 1.f;          // elu(1) == 1 (appended ones column)
  phiU[idx] = v;
}

// ---------------- per-(s,d) posterior solve: w row + KL ----------------
// One workgroup per (d = blockIdx.x, s = blockIdx.y). Single LDS matrix A:
//   lower triangle cycles through  q_prec -> L -> q_cov -> Lq
//   upper triangle holds C = L^{-1} (off-diag), diag of C in dinv[].
__global__ void gi_solve_kernel(const float* __restrict__ phiU,     // [S][M][PUSTR]
                                const float* __restrict__ pmu,      // [dout][M]
                                const float* __restrict__ plogprec, // [dout][M]
                                const float* __restrict__ eps,      // [S][dout][din]
                                float* __restrict__ w_out,          // [S][dout][KP]
                                float* __restrict__ kl_out,         // [S]
                                int din, int KP)
{
  const int d = blockIdx.x, dout = gridDim.x, s = blockIdx.y;
  const int tid = threadIdx.x, T = blockDim.x;
  const int ldA = din + 1;

  extern __shared__ float sm[];
  float* A    = sm;                 // din*ldA
  float* dinv = A + din * ldA;      // 132
  float* vv   = dinv + 132;         // 132
  float* qmu  = vv + 132;           // 132
  float* lpm  = qmu + 132;          // 128
  float* lpv  = lpm + 128;          // 128
  float* red  = lpv + 128;          // T (256)

  const float* pu = phiU + (size_t)s * (M_ * PUSTR);

  if (tid < M_) {
    float lp = expf(plogprec[d * M_ + tid]);
    lpm[tid] = lp;
    lpv[tid] = lp * pmu[d * M_ + tid];
  }
  __syncthreads();

  // q_prec = phiU^T diag(Lp) phiU + I  (lower only), v = phiU^T (Lp*mu)
  for (int idx = tid; idx < din * din; idx += T) {
    int i = idx / din, j = idx - i * din;
    if (j > i) continue;
    float acc = (i == j) ? 1.f : 0.f;
    for (int m = 0; m < M_; ++m) acc += pu[m * PUSTR + i] * lpm[m] * pu[m * PUSTR + j];
    A[i * ldA + j] = acc;
  }
  for (int i = tid; i < din; i += T) {
    float acc = 0.f;
    for (int m = 0; m < M_; ++m) acc += pu[m * PUSTR + i] * lpv[m];
    vv[i] = acc;
  }
  __syncthreads();

  // Cholesky #1: lower(A) <- L, L L^T = q_prec
  for (int k = 0; k < din; ++k) {
    if (tid == 0) A[k * ldA + k] = sqrtf(A[k * ldA + k]);
    __syncthreads();
    float dk = A[k * ldA + k];
    for (int i = k + 1 + tid; i < din; i += T) A[i * ldA + k] /= dk;
    __syncthreads();
    int r = din - k - 1;
    for (int e = tid; e < r * r; e += T) {
      int i = k + 1 + e / r, j = k + 1 + e % r;
      if (j <= i) A[i * ldA + j] -= A[i * ldA + k] * A[j * ldA + k];
    }
    __syncthreads();
  }

  for (int i = tid; i < din; i += T) dinv[i] = 1.f / A[i * ldA + i];
  __syncthreads();

  // C = L^{-1}: column j solved independently by thread j.
  // C[i][j] (i>j) stored in the UPPER triangle at A[j*ldA+i]; C[j][j] = dinv[j].
  if (tid < din) {
    int j = tid;
    for (int i = j + 1; i < din; ++i) {
      float acc = A[i * ldA + j] * dinv[j];
      for (int t = j + 1; t < i; ++t) acc += A[i * ldA + t] * A[j * ldA + t];
      A[j * ldA + i] = -acc * dinv[i];
    }
  }
  __syncthreads();

  // q_cov = C^T C (symmetric): write lower+diag, read only upper+dinv -> race free
  for (int idx = tid; idx < din * din; idx += T) {
    int i = idx / din, j = idx - i * din;
    if (j > i) continue;
    float acc = dinv[i] * ((i == j) ? dinv[i] : A[j * ldA + i]);
    for (int k = i + 1; k < din; ++k) acc += A[i * ldA + k] * A[j * ldA + k];
    A[i * ldA + j] = acc;
  }
  __syncthreads();

  // trace and q_mu = q_cov @ v  (symmetric lower storage: (i,j) -> A[max*ldA+min])
  float tloc = 0.f;
  for (int i = tid; i < din; i += T) {
    tloc += A[i * ldA + i];
    float acc = 0.f;
    for (int j = 0; j < din; ++j) {
      float cij = (j <= i) ? A[i * ldA + j] : A[j * ldA + i];
      acc += cij * vv[j];
    }
    qmu[i] = acc;
  }
  float tr = block_reduce(tloc, red, tid, T);

  float mloc = 0.f;
  for (int i = tid; i < din; i += T) mloc += qmu[i] * qmu[i];
  float musq = block_reduce(mloc, red, tid, T);

  // Cholesky #2: lower(A) <- Lq, Lq Lq^T = q_cov
  for (int k = 0; k < din; ++k) {
    if (tid == 0) A[k * ldA + k] = sqrtf(A[k * ldA + k]);
    __syncthreads();
    float dk = A[k * ldA + k];
    for (int i = k + 1 + tid; i < din; i += T) A[i * ldA + k] /= dk;
    __syncthreads();
    int r = din - k - 1;
    for (int e = tid; e < r * r; e += T) {
      int i = k + 1 + e / r, j = k + 1 + e % r;
      if (j <= i) A[i * ldA + j] -= A[i * ldA + k] * A[j * ldA + k];
    }
    __syncthreads();
  }

  float lloc = 0.f;
  for (int i = tid; i < din; i += T) lloc += logf(A[i * ldA + i]);
  float logdet = 2.f * block_reduce(lloc, red, tid, T);

  // w = q_mu + Lq @ eps ; zero-padded to KP
  const float* ep = eps + ((size_t)s * dout + d) * din;
  for (int i = tid; i < KP; i += T) {
    float val = 0.f;
    if (i < din) {
      float acc = qmu[i];
      for (int j = 0; j <= i; ++j) acc += A[i * ldA + j] * ep[j];
      val = acc;
    }
    w_out[((size_t)s * dout + d) * KP + i] = val;
  }
  if (tid == 0)
    atomicAdd(&kl_out[s], 0.5f * (tr + musq - (float)din - logdet));
}

// ---------------- propagate inducing outputs to next layer's phiU ----------------
__global__ void gi_propagate_kernel(const float* __restrict__ phiU_cur,
                                    const float* __restrict__ w,   // [S][dout][KP]
                                    float* __restrict__ phiU_next,
                                    int din, int dout, int KP)
{
  int idx = blockIdx.x * blockDim.x + threadIdx.x;
  if (idx >= S_ * M_ * PUSTR) return;
  int j = idx % PUSTR;
  int m = (idx / PUSTR) % M_;
  int s = idx / (M_ * PUSTR);
  float v = 0.f;
  if (j < dout) {
    const float* pu = phiU_cur + ((size_t)s * M_ + m) * PUSTR;
    const float* wr = w + ((size_t)s * dout + j) * KP;
    float acc = 0.f;
    for (int i = 0; i < din; ++i) acc += pu[i] * wr[i];
    v = eluf(acc);
  } else if (j == dout) {
    v = 1.f;   // elu(appended 1)
  }
  phiU_next[idx] = v;
}

// ---------------- fused 3-layer forward over N (the hot kernel) ----------------
// 256 threads = 8 waves; each wave owns 16 rows end-to-end. All GEMMs via
// v_wmma_f32_16x16x32_f16; activations staged per-wave in LDS row-major and
// gathered into the ISA fragment layouts (pairs of K -> 32-bit ds loads).
__global__ void gi_forward_kernel(const float* __restrict__ F_in, // [N][32]
                                  const float* __restrict__ w0,   // [S][128][KP0]
                                  const float* __restrict__ w1,   // [S][128][KP1]
                                  const float* __restrict__ w2,   // [S][KP1]
                                  float* __restrict__ F_out,      // [S][N]
                                  int N)
{
  const int s = blockIdx.y;
  const int tid = threadIdx.x;
  const int lane = tid & 31, wave = tid >> 5;
  const int nWaves = blockDim.x >> 5;
  const int rowBase = (blockIdx.x * nWaves + wave) * 16;
  const int TS = PUSTR;

  extern __shared__ char smemraw[];
  _Float16* w0h   = (_Float16*)smemraw;            // [128][KP0]  f16
  _Float16* w1h   = w0h + 128 * KP0;               // [128][KP1]  f16
  float*    w2s   = (float*)(w1h + 128 * KP1);     // [KP1]       f32
  _Float16* tiles = (_Float16*)(w2s + KP1);        // nWaves x [16][TS] f16
  _Float16* tile  = tiles + wave * (16 * TS);

  if (rowBase < N) __builtin_prefetch(F_in + (size_t)rowBase * DIN_, 0, 3);

  // cooperative per-sample weight staging (f32 -> f16), shared by the block
  {
    const float* p0 = w0 + (size_t)s * (128 * KP0);
    for (int i = tid; i < 128 * KP0; i += blockDim.x) w0h[i] = (_Float16)p0[i];
    const float* p1 = w1 + (size_t)s * (128 * KP1);
    for (int i = tid; i < 128 * KP1; i += blockDim.x) w1h[i] = (_Float16)p1[i];
    const float* p2 = w2 + (size_t)s * KP1;
    for (int i = tid; i < KP1; i += blockDim.x) w2s[i] = p2[i];
  }
  __syncthreads();

  // stage layer-0 input: elu(F_in) | 1 | zero-pad to K=64
  for (int idx = lane; idx < 16 * 64; idx += 32) {
    int r = idx >> 6, k = idx & 63;
    int gr = rowBase + r;
    float v = 0.f;
    if (k < DIN_)       { if (gr < N) v = eluf_fast(F_in[(size_t)gr * DIN_ + k]); }
    else if (k == DIN_) v = 1.f;
    tile[r * TS + k] = (_Float16)v;
  }
  asm volatile("s_wait_dscnt 0" ::: "memory");

  union HF { v16h v; unsigned int u[8]; };
  const int arow = lane & 15;
  const int akb  = (lane >> 4) * 8;   // A frag K-base split
  const int bcol = lane & 15;
  const int bkb  = (lane >> 4) * 16;  // unused; load_b uses akb like load_a
  const int crb  = (lane >> 4) * 8;   // C frag row base

  auto load_a = [&](const _Float16* base) {
    HF f;
    const unsigned int* t32 = (const unsigned int*)base;   // base is 4B aligned
#pragma unroll
    for (int j = 0; j < 8; ++j) {
      int k0 = ((j & 4) ? 16 : 0) + akb + 2 * (j & 3);     // K pairs per ISA layout
      f.u[j] = t32[k0 >> 1];
    }
    return f;
  };
  auto load_b = [&](const _Float16* wls, int ldw, int col, int ktbase) {
    HF f;
    const unsigned int* t32 = (const unsigned int*)(wls + (size_t)col * ldw + ktbase);
#pragma unroll
    for (int j = 0; j < 8; ++j) f.u[j] = t32[(((j & 4) ? 16 : 0) + akb + 2 * (j & 3)) >> 1];
    return f;
  };

  // ---- layer 0: [16 x 64] @ w0^T -> [16 x 128], elu, restage ----
  HF a0[2];
#pragma unroll
  for (int kt = 0; kt < 2; ++kt) a0[kt] = load_a(tile + arow * TS + kt * 32);

#pragma unroll
  for (int nt = 0; nt < 8; ++nt) {
    v8f c = {0.f, 0.f, 0.f, 0.f, 0.f, 0.f, 0.f, 0.f};
#pragma unroll
    for (int kt = 0; kt < 2; ++kt) {
      HF b = load_b(w0h, KP0, nt * 16 + bcol, kt * 32);
      c = __builtin_amdgcn_wmma_f32_16x16x32_f16(false, a0[kt].v, false, b.v,
                                                 (short)0, c, false, false);
    }
#pragma unroll
    for (int v = 0; v < 8; ++v)
      tile[(crb + v) * TS + nt * 16 + bcol] = (_Float16)eluf_fast(c[v]);
  }
  // bias column (128) = 1, zero pad 129..159 for layer-1 K=160
  for (int idx = lane; idx < 16 * 32; idx += 32) {
    int r = idx >> 5, k = 128 + (idx & 31);
    tile[r * TS + k] = (_Float16)((k == 128) ? 1.f : 0.f);
  }
  asm volatile("s_wait_dscnt 0" ::: "memory");

  // ---- layer 1: [16 x 160] @ w1^T -> [16 x 128], elu, restage ----
  HF a1[5];
#pragma unroll
  for (int kt = 0; kt < 5; ++kt) a1[kt] = load_a(tile + arow * TS + kt * 32);

#pragma unroll
  for (int nt = 0; nt < 8; ++nt) {
    v8f c = {0.f, 0.f, 0.f, 0.f, 0.f, 0.f, 0.f, 0.f};
#pragma unroll
    for (int kt = 0; kt < 5; ++kt) {
      HF b = load_b(w1h, KP1, nt * 16 + bcol, kt * 32);
      c = __builtin_amdgcn_wmma_f32_16x16x32_f16(false, a1[kt].v, false, b.v,
                                                 (short)0, c, false, false);
    }
#pragma unroll
    for (int v = 0; v < 8; ++v)
      tile[(crb + v) * TS + nt * 16 + bcol] = (_Float16)eluf_fast(c[v]);
  }
  // cols 128..159 still hold (1, 0...) from layer-1 staging -> correct layer-2 input
  asm volatile("s_wait_dscnt 0" ::: "memory");

  // ---- layer 2 (dout = 1): VALU dot per row, packed 32-bit LDS reads ----
  float part = 0.f;
  {
    int r = lane & 15, half = lane >> 4;
    const unsigned int* h32 = (const unsigned int*)(tile + r * TS + half * 80);
    const float* wv = w2s + half * 80;
#pragma unroll
    for (int i = 0; i < 40; ++i) {
      union { unsigned int u; _Float16 h[2]; } p;
      p.u = h32[i];
      part += (float)p.h[0] * wv[2 * i] + (float)p.h[1] * wv[2 * i + 1];
    }
  }
  part += __shfl_down(part, 16);
  {
    int r = lane & 15;
    int gr = rowBase + r;
    if (lane < 16 && gr < N) F_out[(size_t)s * N + gr] = part;
  }
}

// ---------------- host-side launcher ----------------
extern "C" void kernel_launch(void* const* d_in, const int* in_sizes, int n_in,
                              void* d_out, int out_size, void* d_ws, size_t ws_size,
                              hipStream_t stream) {
  (void)in_sizes; (void)n_in; (void)out_size; (void)ws_size;

  const float* F_in = (const float*)d_in[0];
  const float* ip   = (const float*)d_in[1];
  const float* mu0  = (const float*)d_in[2];
  const float* lp0  = (const float*)d_in[3];
  const float* mu1  = (const float*)d_in[4];
  const float* lp1  = (const float*)d_in[5];
  const float* mu2  = (const float*)d_in[6];
  const float* lp2  = (const float*)d_in[7];
  const float* eps0 = (const float*)d_in[8];
  const float* eps1 = (const float*)d_in[9];
  const float* eps2 = (const float*)d_in[10];

  float* out   = (float*)d_out;
  float* F_out = out;                        // [S][N][1] flattened
  float* kl    = out + (size_t)S_ * N_;      // [S]

  // workspace carve-up (floats): 2 phiU ping-pong buffers + 3 weight buffers (~2.3 MB)
  float* ws    = (float*)d_ws;
  float* phiUa = ws;
  float* phiUb = phiUa + (size_t)S_ * M_ * PUSTR;
  float* w0    = phiUb + (size_t)S_ * M_ * PUSTR;
  float* w1    = w0 + (size_t)S_ * H_ * KP0;
  float* w2    = w1 + (size_t)S_ * H_ * KP1;

  const size_t sh33  = (size_t)(33 * 34 + 132 * 3 + 128 * 2 + 256) * sizeof(float);
  const size_t sh129 = (size_t)(129 * 130 + 132 * 3 + 128 * 2 + 256) * sizeof(float);
  const size_t shFwd = (size_t)(128 * KP0 + 128 * KP1) * 2 + KP1 * 4 + (size_t)8 * 16 * PUSTR * 2;

  (void)hipFuncSetAttribute((const void*)gi_solve_kernel,
                            hipFuncAttributeMaxDynamicSharedMemorySize, (int)sh129);
  (void)hipFuncSetAttribute((const void*)gi_forward_kernel,
                            hipFuncAttributeMaxDynamicSharedMemorySize, (int)shFwd);

  zero_kl_kernel<<<1, 32, 0, stream>>>(kl);

  const int tot = S_ * M_ * PUSTR;
  build_phiU0_kernel<<<(tot + 255) / 256, 256, 0, stream>>>(ip, phiUa);

  // layer 0: din=33, dout=128
  gi_solve_kernel<<<dim3(H_, S_), 256, sh33, stream>>>(phiUa, mu0, lp0, eps0, w0, kl, 33, KP0);
  gi_propagate_kernel<<<(tot + 255) / 256, 256, 0, stream>>>(phiUa, w0, phiUb, 33, H_, KP0);

  // layer 1: din=129, dout=128
  gi_solve_kernel<<<dim3(H_, S_), 256, sh129, stream>>>(phiUb, mu1, lp1, eps1, w1, kl, 129, KP1);
  gi_propagate_kernel<<<(tot + 255) / 256, 256, 0, stream>>>(phiUb, w1, phiUa, 129, H_, KP1);

  // layer 2: din=129, dout=1
  gi_solve_kernel<<<dim3(DOUT_, S_), 256, sh129, stream>>>(phiUa, mu2, lp2, eps2, w2, kl, 129, KP1);

  // fused 3-layer forward over all N rows, one sample per blockIdx.y
  const int rowsPerBlock = 8 * 16;
  dim3 grid((N_ + rowsPerBlock - 1) / rowsPerBlock, S_);
  gi_forward_kernel<<<grid, 256, shFwd, stream>>>(F_in, w0, w1, w2, F_out, N_);
}
